// LlamaAttention_58866821759150
// MI455X (gfx1250) — matrix operand order repaired and checked by reading, not compiled, not run on hardware
//
#include <hip/hip_runtime.h>
#include <hip/hip_bf16.h>

typedef _Float16 f16;
typedef __attribute__((ext_vector_type(16))) _Float16 v16h;
typedef __attribute__((ext_vector_type(8)))  float    v8f;
typedef unsigned int u32x4 __attribute__((ext_vector_type(4)));
typedef int          i32x8 __attribute__((ext_vector_type(8)));
typedef int          i32x4 __attribute__((ext_vector_type(4)));

// 16-byte chunk type matching the async-to-LDS builtin's typed pointers
typedef int b128v __attribute__((vector_size(16)));

#define WMMA_F32_F16(A, B, C) \
  __builtin_amdgcn_wmma_f32_16x16x32_f16(false, (A), false, (B), (short)0, (C), false, false)

// ---------------------------------------------------------------------------
// CDNA5 async copy: one 16-byte chunk per lane, global -> LDS, ASYNCcnt.
// ---------------------------------------------------------------------------
__device__ __forceinline__ void async_copy_b128(void* lds_dst, const void* gsrc) {
  __builtin_amdgcn_global_load_async_to_lds_b128(
      (__attribute__((address_space(1))) b128v*)gsrc,
      (__attribute__((address_space(3))) b128v*)lds_dst,
      /*offset=*/0, /*cpol=*/0);
}

__device__ __forceinline__ void wait_asynccnt0() {
  asm volatile("s_wait_asynccnt 0x0" ::: "memory");
}

// ---------------------------------------------------------------------------
// Tensor Data Mover: 2-D f16 tile (rows x cols) from global to LDS, with TDM
// LDS padding of 4 DWORDs (16B) after every 64 DWORDs (256B = one 128-elem
// row), i.e. LDS row stride = cols + 8 halves.  D# layout per ISA §8.3-8.4.
// ---------------------------------------------------------------------------
__device__ __forceinline__ void tdm_load_tile_f16(const f16* gsrc, void* lds_dst,
                                                  int rows, int cols,
                                                  int gstride_elems) {
  typedef __attribute__((address_space(3))) void lds_v;
  const unsigned long long ga = (unsigned long long)gsrc;
  const unsigned lds_off = (unsigned)(unsigned long long)(lds_v*)lds_dst;

  u32x4 g0;
  g0[0] = 1u;                                      // count=1 (valid user D#)
  g0[1] = lds_off;                                 // lds_addr (bytes)
  g0[2] = (unsigned)ga;                            // global_addr[31:0]
  g0[3] = (unsigned)(ga >> 32) | (2u << 30);       // global_addr[56:32] | type=2

  const int w0 = (1 << 16)    // data_size = 2 bytes
               | (1 << 20)    // pad_enable
               | (5 << 22)    // pad_interval: 64 DWORDs
               | (3 << 25);   // pad_amount: 4 DWORDs
  const int w1 = (cols & 0xFFFF) << 16;                            // tensor_dim0 lo16
  const int w2 = ((unsigned)cols >> 16) | ((rows & 0xFFFF) << 16); // dim0 hi / dim1 lo
  const int w3 = ((unsigned)rows >> 16) | ((cols & 0xFFFF) << 16); // dim1 hi / tile_dim0
  const int w4 = rows & 0xFFFF;                                    // tile_dim1 (tile_dim2=0)
  const int w5 = gstride_elems;                                    // tensor_dim0_stride lo32
  i32x8 g1;
  g1[0] = w0; g1[1] = w1; g1[2] = w2; g1[3] = w3;
  g1[4] = w4; g1[5] = w5; g1[6] = 0;  g1[7] = 0;
  i32x4 gz = {0, 0, 0, 0};
  i32x8 gz8 = {0, 0, 0, 0, 0, 0, 0, 0};
  __builtin_amdgcn_tensor_load_to_lds(g0, g1, gz, gz, gz8, 0);
}

// ---------------------------------------------------------------------------
// DPP16 butterfly reductions over a 16-lane half (wave32 DPP row = 16 lanes).
// quad_perm(1,0,3,2)=0xB1, quad_perm(2,3,0,1)=0x4E, row_half_mirror=0x141,
// row_mirror=0x140.  Pure VALU: no LDS traffic, no DScnt stalls.
// ---------------------------------------------------------------------------
template <int CTRL, bool IS_MAX>
__device__ __forceinline__ float dpp16_bfly(float x) {
  const int t = __builtin_amdgcn_update_dpp(0, __builtin_bit_cast(int, x),
                                            CTRL, 0xF, 0xF, true);
  const float y = __builtin_bit_cast(float, t);
  return IS_MAX ? fmaxf(x, y) : (x + y);
}
__device__ __forceinline__ float half16_max(float x) {
  x = dpp16_bfly<0xB1,  true>(x);
  x = dpp16_bfly<0x4E,  true>(x);
  x = dpp16_bfly<0x141, true>(x);
  x = dpp16_bfly<0x140, true>(x);
  return x;
}
__device__ __forceinline__ float half16_sum(float x) {
  x = dpp16_bfly<0xB1,  false>(x);
  x = dpp16_bfly<0x4E,  false>(x);
  x = dpp16_bfly<0x141, false>(x);
  x = dpp16_bfly<0x140, false>(x);
  return x;
}

// ---------------------------------------------------------------------------
// Fragment gathers for 16-bit WMMA operands (wave32): element s holds k = 16*half + s in A and B.
// ---------------------------------------------------------------------------
__device__ __forceinline__ v16h frag_A_rowmajor(const f16* p, int stride, int lane) {
  const int half = (lane >> 4) & 1;
  const f16* row = p + (lane & 15) * stride;
  v16h a;
#pragma unroll
  for (int j = 0; j < 8; ++j) {
    const int kb = 2 * j + (half ? 16 : 0);
    a[2 * j]     = row[kb];
    a[2 * j + 1] = row[kb + 1];
  }
  return a;
}

__device__ __forceinline__ v16h frag_B_rowmajor(const f16* p, int stride, int lane) {
  const int half = (lane >> 4) & 1;
  const int n = lane & 15;
  v16h b;
#pragma unroll
  for (int j = 0; j < 8; ++j) {
    const int k = 2 * j + (half ? 16 : 0);
    b[2 * j]     = p[k * stride + n];
    b[2 * j + 1] = p[(k + 1) * stride + n];
  }
  return b;
}

__device__ __forceinline__ v16h frag_B_colmajor(const f16* p, int stride, int lane) {
  const int half = (lane >> 4) & 1;
  const f16* col = p + (lane & 15) * stride;
  v16h b;
#pragma unroll
  for (int j = 0; j < 8; ++j) {
    const int k = 2 * j + (half ? 16 : 0);
    b[2 * j]     = col[k];
    b[2 * j + 1] = col[k + 1];
  }
  return b;
}

// ---------------------------------------------------------------------------
// fp32 -> f16 bulk convert (one-time; enables raw async b128 staging).
// ---------------------------------------------------------------------------
__global__ void f32_to_f16_kernel(const float* __restrict__ src,
                                  f16* __restrict__ dst, int n4) {
  const int i = blockIdx.x * blockDim.x + threadIdx.x;
  if (i >= n4) return;
  const float4 v = ((const float4*)src)[i];
  typedef __attribute__((ext_vector_type(4))) _Float16 v4h;
  v4h o = {(f16)v.x, (f16)v.y, (f16)v.z, (f16)v.w};
  ((v4h*)dst)[i] = o;
}

// ---------------------------------------------------------------------------
// Tiled GEMM, f16 x f16 -> TOUT.  Block tile 128x64, 8 waves, wave = 32x32.
// Async-to-LDS double buffered K loop (ASYNCcnt).
// ---------------------------------------------------------------------------
template <typename TOUT>
__global__ __launch_bounds__(256) void gemm_wmma_kernel(
    const f16* __restrict__ A, const f16* __restrict__ W, TOUT* __restrict__ C,
    int M, int N, int K) {
  constexpr int BM = 128, BN = 64, BK = 32, PAD = 8;
  constexpr int LA = BK + PAD;   // 40 halves = 80B  (16B multiple)
  constexpr int LB = BN + PAD;   // 72 halves = 144B (16B multiple)
  __shared__ alignas(16) f16 sA[2][BM][LA];
  __shared__ alignas(16) f16 sB[2][BK][LB];

  const int tid  = threadIdx.x;
  const int lane = tid & 31;
  const int w    = tid >> 5;
  const int wm   = w & 3;
  const int wn   = w >> 2;
  const int row0 = blockIdx.x * BM;
  const int col0 = blockIdx.y * BN;

  auto issue_tile = [&](int k0, int buf) {
    for (int c = tid; c < BM * (BK / 8); c += 256) {        // 512 chunks
      const int r = c >> 2, cc = (c & 3) * 8;
      async_copy_b128(&sA[buf][r][cc], &A[(size_t)(row0 + r) * K + k0 + cc]);
    }
    for (int c = tid; c < BK * (BN / 8); c += 256) {        // 256 chunks
      const int r = c >> 3, cc = (c & 7) * 8;
      async_copy_b128(&sB[buf][r][cc], &W[(size_t)(k0 + r) * N + col0 + cc]);
    }
  };

  v8f acc[2][2] = {};
  issue_tile(0, 0);
  int buf = 0;

  for (int k0 = 0; k0 < K; k0 += BK) {
    wait_asynccnt0();        // this wave's fills done
    __syncthreads();         // everyone's fills done / prev compute done
    if (k0 + BK < K) issue_tile(k0 + BK, buf ^ 1);

    const v16h a0 = frag_A_rowmajor(&sA[buf][wm * 32][0],      LA, lane);
    const v16h a1 = frag_A_rowmajor(&sA[buf][wm * 32 + 16][0], LA, lane);
    const v16h b0 = frag_B_rowmajor(&sB[buf][0][wn * 32],      LB, lane);
    const v16h b1 = frag_B_rowmajor(&sB[buf][0][wn * 32 + 16], LB, lane);
    acc[0][0] = WMMA_F32_F16(a0, b0, acc[0][0]);
    acc[0][1] = WMMA_F32_F16(a0, b1, acc[0][1]);
    acc[1][0] = WMMA_F32_F16(a1, b0, acc[1][0]);
    acc[1][1] = WMMA_F32_F16(a1, b1, acc[1][1]);
    buf ^= 1;
  }

  const int half = (lane >> 4) & 1;
  const int cn   = lane & 15;
#pragma unroll
  for (int sm = 0; sm < 2; ++sm)
#pragma unroll
    for (int sn = 0; sn < 2; ++sn)
#pragma unroll
      for (int r = 0; r < 8; ++r) {
        const int rr = row0 + wm * 32 + sm * 16 + r + half * 8;
        const int cc = col0 + wn * 32 + sn * 16 + cn;
        C[(size_t)rr * N + cc] = (TOUT)acc[sm][sn][r];
      }
}

// ---------------------------------------------------------------------------
// RoPE in place on f16: x[d] = x*cos - x[d+64]*sin ; x[d+64] = x[d+64]*cos + x*sin
// ---------------------------------------------------------------------------
__global__ void rope_kernel(f16* __restrict__ x, const float* __restrict__ cosT,
                            const float* __restrict__ sinT, int ncols, int total) {
  const int idx = blockIdx.x * blockDim.x + threadIdx.x;
  if (idx >= total) return;
  const int perS = (ncols >> 7) << 6;
  const int s    = idx / perS;
  const int rem  = idx - s * perS;
  const int h    = rem >> 6;
  const int d    = rem & 63;
  const size_t b = (size_t)s * ncols + h * 128 + d;
  const float x1 = (float)x[b];
  const float x2 = (float)x[b + 64];
  const float c1 = cosT[s * 128 + d],      s1 = sinT[s * 128 + d];
  const float c2 = cosT[s * 128 + d + 64], s2 = sinT[s * 128 + d + 64];
  x[b]      = (f16)(x1 * c1 - x2 * s1);
  x[b + 64] = (f16)(x2 * c2 + x1 * s2);
}

// ---------------------------------------------------------------------------
// Causal flash attention, GQA 16q/4kv, HD=128.  Block = (head, 64-query tile),
// 4 waves x 16 query rows.  K/V tiles fetched by the Tensor Data Mover
// (double buffered, overlapping WMMA), Q tile via async b128 copies.
// Softmax row statistics via DPP16 butterflies (no LDS).
// The 64MB mask input is never read: causality is analytic.
// ---------------------------------------------------------------------------
__global__ __launch_bounds__(128) void flash_attn_kernel(
    const f16* __restrict__ q, const f16* __restrict__ k,
    const f16* __restrict__ v, f16* __restrict__ o) {
  constexpr int NH = 16, NKV = 4, HD = 128, QT = 64, KT = 32, PAD = 8;
  constexpr int LQ = HD + PAD;                 // 136 halves = 272B (16B mult)
  __shared__ alignas(16) f16 sQ[QT][LQ];
  __shared__ alignas(16) f16 sK[2][KT][LQ];
  __shared__ alignas(16) f16 sV[2][KT][LQ];
  __shared__ alignas(16) f16 sP[4][16][KT + PAD];

  const int h    = blockIdx.x;
  const int qt   = blockIdx.y * QT;
  const int kvh  = h / (NH / NKV);
  const int tid  = threadIdx.x;
  const int lane = tid & 31;
  const int w    = tid >> 5;
  const int half = (lane >> 4) & 1;
  const int cn   = lane & 15;

  // Q tile: async b128 copies
  for (int c = tid; c < QT * (HD / 8); c += 128) {          // 1024 chunks
    const int r = c >> 4, cc = (c & 15) * 8;
    async_copy_b128(&sQ[r][cc], &q[(size_t)(qt + r) * (NH * HD) + h * HD + cc]);
  }

  auto tdm_kv = [&](int kb, int buf) {   // wave 0 only; whole 32x128 tiles
    const size_t base = (size_t)(kb * KT) * (NKV * HD) + kvh * HD;
    tdm_load_tile_f16(k + base, &sK[buf][0][0], KT, HD, NKV * HD);
    tdm_load_tile_f16(v + base, &sV[buf][0][0], KT, HD, NKV * HD);
  };

  const int qbase = qt + w * 16;
  const int nkb   = qt / 32 + 2;

  if (w == 0) tdm_kv(0, 0);
  wait_asynccnt0();
  __syncthreads();

  v16h qf[4];
#pragma unroll
  for (int kk = 0; kk < 4; ++kk)
    qf[kk] = frag_A_rowmajor(&sQ[w * 16][kk * 32], LQ, lane);

  v8f acc[8] = {};
  float mrow[8], lrow[8];
#pragma unroll
  for (int r = 0; r < 8; ++r) { mrow[r] = -__builtin_inff(); lrow[r] = 0.f; }

  int buf = 0;
  for (int kb = 0; kb < nkb; ++kb) {
    if (w == 0) __builtin_amdgcn_s_wait_tensorcnt(0);
    __syncthreads();                              // tile[buf] visible to all
    if (kb + 1 < nkb && w == 0) tdm_kv(kb + 1, buf ^ 1);

    v8f s0 = {}, s1 = {};
#pragma unroll
    for (int kk = 0; kk < 4; ++kk) {
      const v16h b0 = frag_B_colmajor(&sK[buf][0][kk * 32],  LQ, lane);
      const v16h b1 = frag_B_colmajor(&sK[buf][16][kk * 32], LQ, lane);
      s0 = WMMA_F32_F16(qf[kk], b0, s0);
      s1 = WMMA_F32_F16(qf[kk], b1, s1);
    }

    const float scale = 0.08838834764831845f;     // 1/sqrt(128)
    float p0[8], p1[8], esc[8];
#pragma unroll
    for (int r = 0; r < 8; ++r) {
      const int row = qbase + r + half * 8;
      float v0 = s0[r] * scale;
      float v1 = s1[r] * scale;
      if (kb * 32 + cn > row)      v0 = -__builtin_inff();
      if (kb * 32 + 16 + cn > row) v1 = -__builtin_inff();
      const float mx   = half16_max(fmaxf(v0, v1));
      const float mnew = fmaxf(mrow[r], mx);
      const float e0 = __expf(v0 - mnew);
      const float e1 = __expf(v1 - mnew);
      const float sum = half16_sum(e0 + e1);
      const float es  = __expf(mrow[r] - mnew);
      lrow[r] = lrow[r] * es + sum;
      mrow[r] = mnew;
      esc[r]  = es;
      p0[r] = e0; p1[r] = e1;
    }
#pragma unroll
    for (int dn = 0; dn < 8; ++dn)
#pragma unroll
      for (int r = 0; r < 8; ++r) acc[dn][r] *= esc[r];

#pragma unroll
    for (int r = 0; r < 8; ++r) {
      sP[w][r + half * 8][cn]      = (f16)p0[r];
      sP[w][r + half * 8][16 + cn] = (f16)p1[r];
    }
    const v16h pf = frag_A_rowmajor(&sP[w][0][0], KT + PAD, lane);
#pragma unroll
    for (int dn = 0; dn < 8; ++dn) {
      const v16h bv = frag_B_rowmajor(&sV[buf][0][dn * 16], LQ, lane);
      acc[dn] = WMMA_F32_F16(pf, bv, acc[dn]);
    }
    buf ^= 1;
  }

#pragma unroll
  for (int dn = 0; dn < 8; ++dn)
#pragma unroll
    for (int r = 0; r < 8; ++r) {
      const int row = qbase + r + half * 8;
      o[(size_t)row * (NH * HD) + h * HD + dn * 16 + cn] =
          (f16)(acc[dn][r] / lrow[r]);
    }
}

// ---------------------------------------------------------------------------
extern "C" void kernel_launch(void* const* d_in, const int* in_sizes, int n_in,
                              void* d_out, int out_size, void* d_ws, size_t ws_size,
                              hipStream_t stream) {
  const float* hid  = (const float*)d_in[0];
  const float* cosT = (const float*)d_in[1];
  const float* sinT = (const float*)d_in[2];
  // d_in[3] = attention_mask: unused (causality computed in-kernel)
  const float* Wq = (const float*)d_in[4];
  const float* Wk = (const float*)d_in[5];
  const float* Wv = (const float*)d_in[6];
  const float* Wo = (const float*)d_in[7];
  float* out = (float*)d_out;

  constexpr int S = 4096, H = 2048, NH = 16, NKV = 4, HD = 128;
  constexpr size_t NQ = (size_t)S * H;          // 8M
  constexpr size_t NKVE = (size_t)S * NKV * HD; // 2M

  f16* qb   = (f16*)d_ws;
  f16* kbuf = qb + NQ;
  f16* vbuf = kbuf + NKVE;
  f16* ab   = vbuf + NKVE;
  f16* hidH = ab + NQ;
  f16* wqH  = hidH + NQ;
  f16* wkH  = wqH + (size_t)H * H;
  f16* wvH  = wkH + (size_t)H * NKV * HD;
  f16* woH  = wvH + (size_t)H * NKV * HD;

  // 0) one-time fp32 -> f16 conversion (halves activation traffic, enables
  //    raw b128 async/TDM staging)
  auto cvt = [&](const float* s, f16* d, size_t n) {
    const int n4 = (int)(n / 4);
    f32_to_f16_kernel<<<(n4 + 255) / 256, 256, 0, stream>>>(s, d, n4);
  };
  cvt(hid, hidH, NQ);
  cvt(Wq, wqH, (size_t)H * H);
  cvt(Wk, wkH, (size_t)H * NKV * HD);
  cvt(Wv, wvH, (size_t)H * NKV * HD);
  cvt(Wo, woH, (size_t)H * H);

  // 1) QKV projections
  gemm_wmma_kernel<f16><<<dim3(S / 128, H / 64), 256, 0, stream>>>(
      hidH, wqH, qb, S, H, H);
  gemm_wmma_kernel<f16><<<dim3(S / 128, (NKV * HD) / 64), 256, 0, stream>>>(
      hidH, wkH, kbuf, S, NKV * HD, H);
  gemm_wmma_kernel<f16><<<dim3(S / 128, (NKV * HD) / 64), 256, 0, stream>>>(
      hidH, wvH, vbuf, S, NKV * HD, H);

  // 2) RoPE
  {
    const int tq = S * NH * 64;
    rope_kernel<<<(tq + 255) / 256, 256, 0, stream>>>(qb, cosT, sinT, NH * HD, tq);
    const int tk = S * NKV * 64;
    rope_kernel<<<(tk + 255) / 256, 256, 0, stream>>>(kbuf, cosT, sinT, NKV * HD, tk);
  }

  // 3) causal flash attention (TDM-fed K/V)
  flash_attn_kernel<<<dim3(NH, S / 64), 128, 0, stream>>>(qb, kbuf, vbuf, ab);

  // 4) output projection -> fp32
  gemm_wmma_kernel<float><<<dim3(S / 128, H / 64), 256, 0, stream>>>(
      ab, woH, out, S, H, H);
}
